// BRISQUE_1580547975367
// MI455X (gfx1250) — matrix operand order repaired and checked by reading, not compiled, not run on hardware
//
#include <hip/hip_runtime.h>
#include <hip/hip_bf16.h>

typedef __attribute__((ext_vector_type(16))) _Float16 v16h;
typedef __attribute__((ext_vector_type(8)))  float    v8f;

union AV { v16h v; _Float16 e[16]; unsigned int u[8]; };
union CV { v8f  v; float    e[8];  };

// bicubic 0.5x polyphase weights: wt[p] = 0.5*cubic(0.5*(4.5-p)), sum == 1 exactly
static constexpr float WT[10] = {
    0.0f, -0.01171875f, -0.03515625f, 0.11328125f, 0.43359375f,
    0.43359375f, 0.11328125f, -0.03515625f, -0.01171875f, 0.0f };

// exact IEEE f16 bit patterns of WT (all are exact binary fractions)
static constexpr unsigned short WTBITS[10] = {
    0x0000, 0xA200, 0xA880, 0x2F40, 0x36F0,
    0x36F0, 0x2F40, 0xA880, 0xA200, 0x0000 };

// Precomputed WMMA A-fragment (banded bicubic matrix, contiguous k order, matched by the B loads):
// element e=2v+h of lane L, chunk c -> K = c*32 + 2*v + (L>=16)*16 + h,
// A[m][k] = wt[k-2m], m = L&15.  Packed two f16 per dword.
struct AmatT { unsigned int w[2][32][8]; };
static constexpr AmatT makeAmat() {
    AmatT t{};
    for (int c = 0; c < 2; ++c)
        for (int lane = 0; lane < 32; ++lane)
            for (int v = 0; v < 8; ++v) {
                unsigned int dw = 0;
                for (int h = 0; h < 2; ++h) {
                    int k = c * 32 + 2 * v +
                            ((lane >= 16) ? 16 : 0) + h;
                    int d = k - 2 * (lane & 15);
                    unsigned int bits = (d >= 0 && d <= 9) ? WTBITS[d] : 0u;
                    dw |= bits << (16 * h);
                }
                t.w[c][lane][v] = dw;
            }
    return t;
}
__device__ __constant__ AmatT c_amat = makeAmat();

__device__ __forceinline__ int mir(int j, int n) {
    int p = j % (2 * n); if (p < 0) p += 2 * n;
    return (p < n) ? p : (2 * n - 1 - p);
}

__device__ __forceinline__ void gauss7(float* g) {
    float s = 0.f;
#pragma unroll
    for (int t = 0; t < 7; ++t) {
        float k = (float)(t - 3) * (6.0f / 7.0f);
        g[t] = __expf(-0.5f * k * k);
        s += g[t];
    }
    float inv = 1.0f / s;
#pragma unroll
    for (int t = 0; t < 7; ++t) g[t] *= inv;
}

// ---------------- MSCN: fused separable 7-tap Gaussian (zero padded) ---------
// block 256 threads -> 32x32 output tile, halo 3; grid tiles DIM exactly
template <int DIM>
__global__ void mscn_kernel(const float* __restrict__ src, float* __restrict__ dst) {
    __shared__ float xs[38][40];
    __shared__ float hx[38][33];
    __shared__ float hx2[38][33];
    const int r0 = blockIdx.y * 32, c0 = blockIdx.x * 32;
    const int tid = threadIdx.x;
    float g[7]; gauss7(g);

    for (int i = tid; i < 38 * 40; i += 256) {
        int r = i / 40, c = i % 40;
        int gr = r0 + r - 3, gc = c0 + c - 3;
        float v = 0.f;
        if (c < 38 && gr >= 0 && gr < DIM && gc >= 0 && gc < DIM)
            v = src[(long)gr * DIM + gc];
        xs[r][c] = v;
    }
    __syncthreads();
    for (int i = tid; i < 38 * 32; i += 256) {
        int r = i / 32, c = i % 32;
        float a = 0.f, b = 0.f;
#pragma unroll
        for (int t = 0; t < 7; ++t) { float v = xs[r][c + t]; a += g[t] * v; b += g[t] * v * v; }
        hx[r][c] = a; hx2[r][c] = b;
    }
    __syncthreads();
    for (int i = tid; i < 32 * 32; i += 256) {
        int r = i / 32, c = i % 32;
        float mu = 0.f, e2 = 0.f;
#pragma unroll
        for (int t = 0; t < 7; ++t) { mu += g[t] * hx[r + t][c]; e2 += g[t] * hx2[r + t][c]; }
        float x = xs[r + 3][c + 3];
        float sig = sqrtf(fabsf(e2 - mu * mu));
        dst[(long)(r0 + r) * DIM + (c0 + c)] = (x - mu) / (sig + 1.0f);
    }
}

// ------------- bicubic 0.5x downsample, vertical pass via WMMA ---------------
// block 128 (4 waves) -> 32x32 output tile; input 72x72 with mirror boundary.
// vertical pass per 16x16 quadrant: D = A(16x64 banded f16) x B(64x16 f16), f32 acc
// fixed geometry: 4096 -> 2048, grid 64x64 tiles exactly (no store guards)
__global__ void down_kernel(const float* __restrict__ src, float* __restrict__ dst) {
    constexpr int N = 4096, M = 2048;
    __shared__ float    xs[72][73];
    __shared__ _Float16 tbT[32][96];   // [out-col][tmp-row]; rows 72..95 kept zero
    const int ro0 = blockIdx.y * 32, co0 = blockIdx.x * 32;
    const int ri0 = 2 * ro0 - 4,     ci0 = 2 * co0 - 4;
    const int tid = threadIdx.x;

    for (int i = tid; i < 72 * 72; i += 128) {
        int r = i / 72, c = i % 72;
        xs[r][c] = src[(long)mir(ri0 + r, N) * N + mir(ci0 + c, N)];
    }
    __syncthreads();
    // horizontal polyphase pass, stored transposed for contiguous WMMA-B loads
    for (int i = tid; i < 32 * 96; i += 128) {
        int n = i / 96, r = i % 96;
        float s = 0.f;
        if (r < 72) {
#pragma unroll
            for (int p = 1; p <= 8; ++p) s += WT[p] * xs[r][2 * n + p];
        }
        tbT[n][r] = (_Float16)s;
    }
    __syncthreads();

    const int wv = tid >> 5, lane = tid & 31;
    const int mr  = (wv >> 1) * 16;      // quadrant row base (out-local)
    const int nc2 = (wv & 1) * 16;       // quadrant col base (out-local)
    const int khalf16 = (lane >= 16) ? 16 : 0;
    const int bcol = nc2 + (lane & 15);

    // branch-free A fragments from the compile-time constant table
    AV a0, a1;
#pragma unroll
    for (int q = 0; q < 8; ++q) { a0.u[q] = c_amat.w[0][lane][q]; a1.u[q] = c_amat.w[1][lane][q]; }

    CV acc; acc.v = (v8f){0.f, 0.f, 0.f, 0.f, 0.f, 0.f, 0.f, 0.f};
    // chunk 0: K = 0..31
    {
        const v16h* bp = (const v16h*)&tbT[bcol][2 * mr + khalf16];
        acc.v = __builtin_amdgcn_wmma_f32_16x16x32_f16(false, a0.v, false, *bp,
                                                       (short)0, acc.v, false, false);
    }
    // chunk 1: K = 32..63 (rows >= 72 are the zero pad, matching A's zero band)
    {
        const v16h* bp = (const v16h*)&tbT[bcol][2 * mr + 32 + khalf16];
        acc.v = __builtin_amdgcn_wmma_f32_16x16x32_f16(false, a1.v, false, *bp,
                                                       (short)0, acc.v, false, false);
    }
    // D layout: VGPR r -> M = r + 8*(lane>=16), N = lane&15.
    // Grid tiles 2048^2 exactly -> unconditional stores, one base + imm offsets.
    {
        const int gr0 = ro0 + mr + ((lane >= 16) ? 8 : 0);
        float* p = dst + (long)gr0 * M + (co0 + bcol);
#pragma unroll
        for (int r = 0; r < 8; ++r) p[(long)r * M] = acc.e[r];
    }
}

// ------------- AGGD moment sums: 5 signals x 4 scalars, deterministic --------
__global__ void stat_kernel(const float* __restrict__ m, int H, int W, int lw,
                            float* __restrict__ part) {
    const int tid = threadIdx.x;
    const long total = (long)H * W;
    float loc[20];
#pragma unroll
    for (int k = 0; k < 20; ++k) loc[k] = 0.f;

    const int wm = W - 1, hm = H - 1;
    const long stride = (long)gridDim.x * 256;
    for (long idx = (long)blockIdx.x * 256 + tid; idx < total; idx += stride) {
        __builtin_prefetch(&m[(idx + stride) & (total - 1)], 0, 1);
        int i = (int)(idx >> lw), j = (int)(idx & wm);
        float mv = m[idx];
        int jm1 = (j - 1) & wm, im1 = (i - 1) & hm, jp1 = (j + 1) & wm;
        float Lf = m[(long)i * W + jm1];
        float Uf = m[(long)im1 * W + j];
        float ULf = m[(long)im1 * W + jm1];
        float URf = m[(long)im1 * W + jp1];
        float s[5] = {mv, mv * Lf, mv * Uf, mv * ULf, mv * URf};
#pragma unroll
        for (int k = 0; k < 5; ++k) {
            float v = s[k], v2 = v * v;
            bool n = v < 0.f;
            loc[4 * k + 0] += n ? 1.f : 0.f;
            loc[4 * k + 1] += n ? v2 : 0.f;
            loc[4 * k + 2] += n ? 0.f : v2;
            loc[4 * k + 3] += fabsf(v);
        }
    }
    __shared__ float red[256];
    for (int k = 0; k < 20; ++k) {
        red[tid] = loc[k];
        __syncthreads();
        for (int s = 128; s > 0; s >>= 1) {
            if (tid < s) red[tid] += red[tid + s];
            __syncthreads();
        }
        if (tid == 0) part[(long)blockIdx.x * 20 + k] = red[0];
        __syncthreads();
    }
}

__global__ void finalize_kernel(const float* __restrict__ p1, int nb1,
                                const float* __restrict__ p2, int nb2,
                                double* __restrict__ acc) {
    int t = threadIdx.x;
    if (t < 20) {
        double s = 0.0;
        for (int b = 0; b < nb1; ++b) s += (double)p1[(long)b * 20 + t];
        acc[t] = s;
    } else if (t < 40) {
        int k = t - 20;
        double s = 0.0;
        for (int b = 0; b < nb2; ++b) s += (double)p2[(long)b * 20 + k];
        acc[20 + k] = s;
    }
}

// ------------- GGD table search (on-the-fly Gamma) + feature assembly --------
__global__ void feat_kernel(const double* __restrict__ acc, float* __restrict__ out) {
    const int tid = threadIdx.x;
    const int w = tid >> 5, lane = tid & 31;
    if (w >= 10) return;
    const int map = w / 5, sig = w % 5;
    const double* a4 = acc + map * 20 + sig * 4;
    const double cl = a4[0], sneg = a4[1], spos = a4[2], sabs = a4[3];
    const double Ntot = (map == 0) ? 4096.0 * 4096.0 : 2048.0 * 2048.0;
    const double cr = Ntot - cl;

    const float lms = sqrtf((float)(sneg / fmax(cl, 1.0)));
    const float rms = sqrtf((float)(spos / fmax(cr, 1.0)));
    const float gh = lms / rms;
    const float meanabs = (float)(sabs / Ntot);
    const float meansq  = (float)((sneg + spos) / Ntot);
    const float rhat = meanabs * meanabs / meansq;
    const float gh2 = gh * gh;
    const float rnorm = rhat * (gh * gh2 + 1.0f) * (gh + 1.0f) / ((gh2 + 1.0f) * (gh2 + 1.0f));

    float bd = 1e30f; int bi = 0;
    for (int i = lane; i < 9800; i += 32) {
        float al = 0.2f + 0.001f * (float)i;
        float r = expf(2.f * lgammaf(2.f / al) - lgammaf(1.f / al) - lgammaf(3.f / al));
        float d = fabsf(r - rnorm);
        if (d < bd || (d == bd && i < bi)) { bd = d; bi = i; }
    }
#pragma unroll
    for (int off = 16; off > 0; off >>= 1) {
        float od = __shfl_down(bd, off, 32);
        int   oi = __shfl_down(bi, off, 32);
        if (od < bd || (od == bd && oi < bi)) { bd = od; bi = oi; }
    }
    if (lane == 0) {
        float alpha = 0.2f + 0.001f * (float)bi;
        float g1 = expf(lgammaf(1.f / alpha));
        float g2 = expf(lgammaf(2.f / alpha));
        float g3 = expf(lgammaf(3.f / alpha));
        float ratio = sqrtf(g1) / sqrtf(g3);
        float bl = ratio * lms, br = ratio * rms;
        float Nf = (br - bl) * (g2 / g1);
        int base = map * 18;
        if (sig == 0) {
            out[base + 0] = alpha;
            out[base + 1] = 0.5f * (bl + br);
        } else {
            int o = base + 2 + (sig - 1) * 4;
            out[o + 0] = alpha; out[o + 1] = Nf; out[o + 2] = bl; out[o + 3] = br;
        }
    }
}

extern "C" void kernel_launch(void* const* d_in, const int* in_sizes, int n_in,
                              void* d_out, int out_size, void* d_ws, size_t ws_size,
                              hipStream_t stream) {
    (void)in_sizes; (void)n_in; (void)out_size; (void)ws_size;
    const float* x = (const float*)d_in[0];
    float* out = (float*)d_out;

    char* ws = (char*)d_ws;
    double* acc  = (double*)ws;                       // 40 doubles
    float* m1    = (float*)(ws + 512);                // 4096^2
    float* dhalf = m1 + 4096UL * 4096UL;              // 2048^2
    float* m2    = dhalf + 2048UL * 2048UL;           // 2048^2
    float* part1 = m2 + 2048UL * 2048UL;              // 2048*20
    float* part2 = part1 + 2048UL * 20UL;             // 1024*20

    mscn_kernel<4096><<<dim3(128, 128), 256, 0, stream>>>(x, m1);
    down_kernel<<<dim3(64, 64), 128, 0, stream>>>(x, dhalf);
    mscn_kernel<2048><<<dim3(64, 64), 256, 0, stream>>>(dhalf, m2);
    stat_kernel<<<2048, 256, 0, stream>>>(m1, 4096, 4096, 12, part1);
    stat_kernel<<<1024, 256, 0, stream>>>(m2, 2048, 2048, 11, part2);
    finalize_kernel<<<1, 64, 0, stream>>>(part1, 2048, part2, 1024, acc);
    feat_kernel<<<1, 320, 0, stream>>>(acc, out);
}
